// TokenLevelMAW_13168369729819
// MI455X (gfx1250) — matrix operand order repaired and checked by reading, not compiled, not run on hardware
//
#include <hip/hip_runtime.h>
#include <hip/hip_bf16.h>

typedef __attribute__((ext_vector_type(16))) __bf16 v16bf;
typedef __attribute__((ext_vector_type(8)))  __bf16 v8bf;
typedef __attribute__((ext_vector_type(8)))  float  v8f;

#define BB   2
#define SS   2048
#define HH   1024
#define NHH  16
#define HD   64
#define LN_EPS 1e-5f

union U16 { v16bf v; v8bf h[2]; __bf16 e[16]; };

// ---- WMMA tile helpers (wave32 operand layouts per CDNA5 ISA) ----
// A (16x32 bf16): lane m = lane&15; halves at kc + 8*hi + (0..7) and +16.
__device__ __forceinline__ void load_a_tiles(const __bf16* __restrict__ A, int K,
                                             int mBase, int lm, int hi, int kc,
                                             U16 (&au)[2]) {
    #pragma unroll
    for (int i = 0; i < 2; ++i) {
        const __bf16* arow = A + (size_t)(mBase + i * 16 + lm) * K + kc + 8 * hi;
        au[i].h[0] = *(const v8bf*)(arow);
        au[i].h[1] = *(const v8bf*)(arow + 16);
    }
}
// B (32x16 bf16): lane col n = nBase + t*16 + (lane&15); halves at kc + 8*hi + (0..7) and +16 in W[N][K].
__device__ __forceinline__ void load_b_tiles(const __bf16* __restrict__ W, int K,
                                             int nBase, int lm, int hi, int kc,
                                             U16 (&bu)[4]) {
    #pragma unroll
    for (int t = 0; t < 4; ++t) {
        const __bf16* wrow = W + (size_t)(nBase + t * 16 + lm) * K + kc + 8 * hi;
        bu[t].h[0] = *(const v8bf*)(wrow);
        bu[t].h[1] = *(const v8bf*)(wrow + 16);
    }
}
__device__ __forceinline__ void mma8(const U16 (&au)[2], const U16 (&bu)[4], v8f (&acc)[2][4]) {
    #pragma unroll
    for (int t = 0; t < 4; ++t)
        #pragma unroll
        for (int i = 0; i < 2; ++i)
            acc[i][t] = __builtin_amdgcn_wmma_f32_16x16x32_bf16(
                false, au[i].v, false, bu[t].v, (short)0, acc[i][t], false, false);
}
__device__ __forceinline__ void copy_tiles(U16 (&da)[2], U16 (&db)[4],
                                           const U16 (&sa)[2], const U16 (&sb)[4]) {
    #pragma unroll
    for (int i = 0; i < 2; ++i) da[i].v = sa[i].v;
    #pragma unroll
    for (int t = 0; t < 4; ++t) db[t].v = sb[t].v;
}

// ---------------------------------------------------------------------------
// f32 -> bf16 elementwise cast
// ---------------------------------------------------------------------------
__global__ void cast_f32_bf16(const float* __restrict__ in, __bf16* __restrict__ out, int n) {
    int i = blockIdx.x * blockDim.x + threadIdx.x;
    if (i < n) out[i] = (__bf16)in[i];
}

// ---------------------------------------------------------------------------
// C[M,N] = A[M,K](bf16) @ W[N,K](bf16)^T + bias   (f32 out, optional bf16 out)
// grid: (M/128, N/64), block 128 = 4 waves; wave: 32(M) x 64(N).
// Double-buffered K-chunks; __launch_bounds__(128,1) gives the allocator the
// full VGPR budget so the pipeline lives entirely in registers (no spills).
// ---------------------------------------------------------------------------
__global__ void __launch_bounds__(128, 1)
gemm_bf16_nt(const __bf16* __restrict__ A, const __bf16* __restrict__ W,
             const float* __restrict__ bias,
             float* __restrict__ C, __bf16* __restrict__ Cb,
             int M, int N, int K) {
    const int wave = threadIdx.x >> 5;
    const int lane = threadIdx.x & 31;
    const int hi   = lane >> 4;      // 0 | 1
    const int lm   = lane & 15;
    const int mBase = blockIdx.x * 128 + wave * 32;
    const int nBase = blockIdx.y * 64;

    v8f acc[2][4] = {};
    U16 au[2], bu[4], an[2], bn[4];

    load_a_tiles(A, K, mBase, lm, hi, 0, au);
    load_b_tiles(W, K, nBase, lm, hi, 0, bu);

    for (int kc = 0; kc < K - 32; kc += 32) {
        // issue next chunk's loads first -> latency hidden behind 8 WMMAs
        load_a_tiles(A, K, mBase, lm, hi, kc + 32, an);
        load_b_tiles(W, K, nBase, lm, hi, kc + 32, bn);
        __builtin_prefetch(A + (size_t)(mBase + lm) * K + kc + 64, 0, 1);
        mma8(au, bu, acc);
        copy_tiles(au, bu, an, bn);
    }
    mma8(au, bu, acc);

    // ---- epilogue: c[r] = C[m = r + 8*hi][n = lm] within each 16x16 tile
    #pragma unroll
    for (int i = 0; i < 2; ++i) {
        #pragma unroll
        for (int t = 0; t < 4; ++t) {
            const int col = nBase + t * 16 + lm;
            const float bv = bias[col];
            #pragma unroll
            for (int r = 0; r < 8; ++r) {
                const int row = mBase + i * 16 + r + 8 * hi;
                const float val = acc[i][t][r] + bv;
                C[(size_t)row * N + col] = val;
                if (Cb) Cb[(size_t)row * N + col] = (__bf16)val;
            }
        }
    }
}

// ---------------------------------------------------------------------------
// qm/km: per-(b,h,s) mean over head-dim of f32 activations laid out (B,S,H)
// ---------------------------------------------------------------------------
__global__ void head_mean(const float* __restrict__ x, float* __restrict__ out) {
    int idx = blockIdx.x * blockDim.x + threadIdx.x;       // idx over B*NH*S
    if (idx >= BB * NHH * SS) return;
    int s  = idx % SS;
    int bh = idx / SS;
    int b  = bh / NHH;
    int h  = bh % NHH;
    const float* p = x + ((size_t)b * SS + s) * HH + h * HD;
    float acc = 0.f;
    #pragma unroll
    for (int d = 0; d < HD; ++d) acc += p[d];
    out[idx] = acc * (1.0f / HD);
}

// ---------------------------------------------------------------------------
// per-(b,h): max/min of km over valid (masked) k
// ---------------------------------------------------------------------------
__global__ void km_minmax(const float* __restrict__ km, const float* __restrict__ mask,
                          float* __restrict__ kmax, float* __restrict__ kmin) {
    const int bh = blockIdx.x;
    const int b  = bh / NHH;
    float mx = -3.0e38f, mn = 3.0e38f;
    for (int s = threadIdx.x; s < SS; s += blockDim.x) {
        if (mask[(size_t)b * SS + s] > 0.f) {
            float v = km[(size_t)bh * SS + s];
            mx = fmaxf(mx, v);
            mn = fminf(mn, v);
        }
    }
    __shared__ float rmx[256], rmn[256];
    rmx[threadIdx.x] = mx; rmn[threadIdx.x] = mn;
    __syncthreads();
    for (int off = 128; off > 0; off >>= 1) {
        if (threadIdx.x < off) {
            rmx[threadIdx.x] = fmaxf(rmx[threadIdx.x], rmx[threadIdx.x + off]);
            rmn[threadIdx.x] = fminf(rmn[threadIdx.x], rmn[threadIdx.x + off]);
        }
        __syncthreads();
    }
    if (threadIdx.x == 0) { kmax[bh] = rmx[0]; kmin[bh] = rmn[0]; }
}

// ---------------------------------------------------------------------------
// vT[b,h,d,k] (bf16) = vF[b,k,h*64+d] (f32)  -- K-contiguous B operand layout
// ---------------------------------------------------------------------------
__global__ void transpose_v(const float* __restrict__ vF, __bf16* __restrict__ vT) {
    size_t idx = (size_t)blockIdx.x * blockDim.x + threadIdx.x;  // over B*NH*HD*S
    if (idx >= (size_t)BB * NHH * HD * SS) return;
    int k  = idx % SS;
    size_t r = idx / SS;
    int d  = r % HD;
    int bh = r / HD;
    int b  = bh / NHH, h = bh % NHH;
    vT[idx] = (__bf16)vF[((size_t)b * SS + k) * HH + h * HD + d];
}

// ---------------------------------------------------------------------------
// Fused rank-1 softmax attention + (w @ v) via WMMA.
// km/mask rows staged to LDS once via global_load_async_to_lds_b128 (ASYNCcnt);
// hot loop is barrier-free and double-buffers the vT B-tiles.
// grid: (S/128, NH, B), block 128 = 4 waves; wave: 32 q-rows x 64 d-cols.
// ---------------------------------------------------------------------------
__global__ void __launch_bounds__(128, 1)
attn_ctx(const float* __restrict__ qm, const float* __restrict__ km,
         const float* __restrict__ kmax, const float* __restrict__ kmin,
         const float* __restrict__ mask,
         const __bf16* __restrict__ vT, __bf16* __restrict__ ctxB) {
    const int b = blockIdx.z, h = blockIdx.y;
    const int bh = b * NHH + h;
    const int wave = threadIdx.x >> 5;
    const int lane = threadIdx.x & 31;
    const int hi   = lane >> 4;
    const int lm   = lane & 15;
    const int qBase = blockIdx.x * 128 + wave * 32;

    __shared__ __align__(16) float kmS[SS];
    __shared__ __align__(16) float maskS[SS];
    __shared__ float zred[256];

    // ---- async prologue: stage km row + mask row into LDS (16 floats/thread)
    {
        const float* gkm = km   + (size_t)bh * SS;
        const float* gmk = mask + (size_t)b  * SS;
        const int base = threadIdx.x * 16;
        #pragma unroll
        for (int i = 0; i < 4; ++i) {
            const float* pk = gkm + base + i * 4;
            const float* pm = gmk + base + i * 4;
            unsigned lds_k = (unsigned)(size_t)(&kmS[base + i * 4]);
            unsigned lds_m = (unsigned)(size_t)(&maskS[base + i * 4]);
            asm volatile("global_load_async_to_lds_b128 %0, %1, off"
                         :: "v"(lds_k), "v"(pk) : "memory");
            asm volatile("global_load_async_to_lds_b128 %0, %1, off"
                         :: "v"(lds_m), "v"(pm) : "memory");
        }
        asm volatile("s_wait_asynccnt 0x0" ::: "memory");
    }
    __syncthreads();

    float myQm[2], mMax[2], zacc[2] = {0.f, 0.f};
    #pragma unroll
    for (int i = 0; i < 2; ++i) {
        myQm[i] = qm[(size_t)bh * SS + qBase + i * 16 + lm];
        mMax[i] = (myQm[i] >= 0.f) ? myQm[i] * kmax[bh] : myQm[i] * kmin[bh];
    }

    v8f acc[2][4] = {};
    const __bf16* vTh = vT + (size_t)bh * HD * SS;

    // E-tile generator: A-layout element (m=lm, k = kc + 8*hi + j | +16)
    auto gen_e = [&](int kc, U16 (&eu)[2]) {
        #pragma unroll
        for (int i = 0; i < 2; ++i) {
            #pragma unroll
            for (int j = 0; j < 8; ++j) {
                const int k0 = kc + 8 * hi + j;
                const int k1 = k0 + 16;
                const float e0 = (maskS[k0] > 0.f) ? __expf(myQm[i] * kmS[k0] - mMax[i]) : 0.f;
                const float e1 = (maskS[k1] > 0.f) ? __expf(myQm[i] * kmS[k1] - mMax[i]) : 0.f;
                zacc[i] += e0 + e1;
                eu[i].e[j]     = (__bf16)e0;
                eu[i].e[8 + j] = (__bf16)e1;
            }
        }
    };

    U16 bu[4], bn[4], eu[2];
    load_b_tiles(vTh, SS, 0, lm, hi, 0, bu);

    for (int kc = 0; kc < SS - 32; kc += 32) {
        // next chunk's vT loads issued first; exp generation overlaps them
        load_b_tiles(vTh, SS, 0, lm, hi, kc + 32, bn);
        __builtin_prefetch(vTh + (size_t)lm * SS + kc + 64, 0, 1);
        gen_e(kc, eu);
        #pragma unroll
        for (int t = 0; t < 4; ++t)
            #pragma unroll
            for (int i = 0; i < 2; ++i)
                acc[i][t] = __builtin_amdgcn_wmma_f32_16x16x32_bf16(
                    false, eu[i].v, false, bu[t].v, (short)0, acc[i][t], false, false);
        #pragma unroll
        for (int t = 0; t < 4; ++t) bu[t].v = bn[t].v;
    }
    gen_e(SS - 32, eu);
    #pragma unroll
    for (int t = 0; t < 4; ++t)
        #pragma unroll
        for (int i = 0; i < 2; ++i)
            acc[i][t] = __builtin_amdgcn_wmma_f32_16x16x32_bf16(
                false, eu[i].v, false, bu[t].v, (short)0, acc[i][t], false, false);

    // ---- Z reduction: per tile, rows m shared by lane m and lane m+16
    zred[wave * 64 + 0 * 32 + lane] = zacc[0];
    zred[wave * 64 + 1 * 32 + lane] = zacc[1];
    __syncthreads();

    #pragma unroll
    for (int i = 0; i < 2; ++i) {
        #pragma unroll
        for (int t = 0; t < 4; ++t) {
            #pragma unroll
            for (int r = 0; r < 8; ++r) {
                const int m = r + 8 * hi;                       // C/D layout row
                const float Zr = zred[wave * 64 + i * 32 + m] +
                                 zred[wave * 64 + i * 32 + 16 + m];
                const float val = (Zr > 0.f) ? acc[i][t][r] / Zr : 0.f;  // NaN->0 path
                const int q = qBase + i * 16 + m;
                const int d = t * 16 + lm;
                ctxB[((size_t)b * SS + q) * HH + h * HD + d] = (__bf16)val;
            }
        }
    }
}

// ---------------------------------------------------------------------------
// out = LayerNorm(hidden + proj) * ln_w + ln_b ; one block (256 thr) per row
// ---------------------------------------------------------------------------
__global__ void residual_ln(const float* __restrict__ hidden, const float* __restrict__ proj,
                            const float* __restrict__ lnw, const float* __restrict__ lnb,
                            float* __restrict__ out) {
    const int row = blockIdx.x;
    const float* hrow = hidden + (size_t)row * HH;
    const float* prow = proj   + (size_t)row * HH;

    float x[4];
    float s = 0.f;
    #pragma unroll
    for (int i = 0; i < 4; ++i) {
        const int c = threadIdx.x + i * 256;
        x[i] = hrow[c] + prow[c];
        s += x[i];
    }
    __shared__ float red[256];
    red[threadIdx.x] = s;
    __syncthreads();
    for (int off = 128; off > 0; off >>= 1) {
        if (threadIdx.x < off) red[threadIdx.x] += red[threadIdx.x + off];
        __syncthreads();
    }
    const float mu = red[0] * (1.0f / HH);
    __syncthreads();

    float v = 0.f;
    #pragma unroll
    for (int i = 0; i < 4; ++i) { const float d = x[i] - mu; v += d * d; }
    red[threadIdx.x] = v;
    __syncthreads();
    for (int off = 128; off > 0; off >>= 1) {
        if (threadIdx.x < off) red[threadIdx.x] += red[threadIdx.x + off];
        __syncthreads();
    }
    const float inv = rsqrtf(red[0] * (1.0f / HH) + LN_EPS);
    #pragma unroll
    for (int i = 0; i < 4; ++i) {
        const int c = threadIdx.x + i * 256;
        out[(size_t)row * HH + c] = (x[i] - mu) * inv * lnw[c] + lnb[c];
    }
}

// ---------------------------------------------------------------------------
extern "C" void kernel_launch(void* const* d_in, const int* in_sizes, int n_in,
                              void* d_out, int out_size, void* d_ws, size_t ws_size,
                              hipStream_t stream) {
    const float* hidden = (const float*)d_in[0];
    const float* amask  = (const float*)d_in[1];
    const float* Wq = (const float*)d_in[2];  const float* bq = (const float*)d_in[3];
    const float* Wk = (const float*)d_in[4];  const float* bk = (const float*)d_in[5];
    const float* Wv = (const float*)d_in[6];  const float* bv = (const float*)d_in[7];
    const float* Wo = (const float*)d_in[8];  const float* bo = (const float*)d_in[9];
    // d_in[10..13]: policy MLP -- provably dead code (one_hot(...).sum(-1) == 1)
    const float* lnw = (const float*)d_in[14];
    const float* lnb = (const float*)d_in[15];
    float* out = (float*)d_out;

    const size_t nTok  = (size_t)BB * SS;          // 4096
    const size_t nHid  = nTok * HH;                // 4,194,304
    const size_t nW    = (size_t)HH * HH;          // 1,048,576

    // ---- workspace carve-up (all offsets 256B aligned)
    char* base = (char*)d_ws;
    size_t off = 0;
    auto carve = [&](size_t bytes) { void* p = base + off; off += (bytes + 255) & ~(size_t)255; return p; };
    __bf16* hiddenB = (__bf16*)carve(nHid * 2);
    __bf16* WqB = (__bf16*)carve(nW * 2);
    __bf16* WkB = (__bf16*)carve(nW * 2);
    __bf16* WvB = (__bf16*)carve(nW * 2);
    __bf16* WoB = (__bf16*)carve(nW * 2);
    float*  qF  = (float*)carve(nHid * 4);
    float*  kF  = (float*)carve(nHid * 4);
    float*  vF  = (float*)carve(nHid * 4);
    float*  qmP = (float*)carve((size_t)BB * NHH * SS * 4);
    float*  kmP = (float*)carve((size_t)BB * NHH * SS * 4);
    float*  kmaxP = (float*)carve(BB * NHH * 4);
    float*  kminP = (float*)carve(BB * NHH * 4);
    __bf16* vT   = (__bf16*)carve(nHid * 2);
    __bf16* ctxB = (__bf16*)carve(nHid * 2);
    float*  projF = (float*)carve(nHid * 4);
    (void)ws_size; (void)in_sizes; (void)n_in; (void)out_size;

    // ---- 0) casts to bf16
    cast_f32_bf16<<<dim3((nHid + 255) / 256), dim3(256), 0, stream>>>(hidden, hiddenB, (int)nHid);
    cast_f32_bf16<<<dim3((nW + 255) / 256), dim3(256), 0, stream>>>(Wq, WqB, (int)nW);
    cast_f32_bf16<<<dim3((nW + 255) / 256), dim3(256), 0, stream>>>(Wk, WkB, (int)nW);
    cast_f32_bf16<<<dim3((nW + 255) / 256), dim3(256), 0, stream>>>(Wv, WvB, (int)nW);
    cast_f32_bf16<<<dim3((nW + 255) / 256), dim3(256), 0, stream>>>(Wo, WoB, (int)nW);

    // ---- 1) QKV projections (WMMA bf16), 32x64 tiles per wave, pipelined
    dim3 ggrid(nTok / 128, HH / 64), gblk(128);
    gemm_bf16_nt<<<ggrid, gblk, 0, stream>>>(hiddenB, WqB, bq, qF, ((__bf16*)nullptr), (int)nTok, HH, HH);
    gemm_bf16_nt<<<ggrid, gblk, 0, stream>>>(hiddenB, WkB, bk, kF, ((__bf16*)nullptr), (int)nTok, HH, HH);
    gemm_bf16_nt<<<ggrid, gblk, 0, stream>>>(hiddenB, WvB, bv, vF, ((__bf16*)nullptr), (int)nTok, HH, HH);

    // ---- 2) head means, masked km min/max, v transpose
    head_mean<<<dim3((BB * NHH * SS) / 256), dim3(256), 0, stream>>>(qF, qmP);
    head_mean<<<dim3((BB * NHH * SS) / 256), dim3(256), 0, stream>>>(kF, kmP);
    km_minmax<<<dim3(BB * NHH), dim3(256), 0, stream>>>(kmP, amask, kmaxP, kminP);
    transpose_v<<<dim3((nHid + 255) / 256), dim3(256), 0, stream>>>(vF, vT);

    // ---- 3) fused rank-1 softmax + (w @ v) via WMMA + async-LDS staging
    attn_ctx<<<dim3(SS / 128, NHH, BB), dim3(128), 0, stream>>>(qmP, kmP, kmaxP, kminP, amask, vT, ctxB);

    // ---- 4) output projection (WMMA bf16)
    gemm_bf16_nt<<<ggrid, gblk, 0, stream>>>(ctxB, WoB, bo, projF, ((__bf16*)nullptr), (int)nTok, HH, HH);

    // ---- 5) residual + LayerNorm
    residual_ln<<<dim3(nTok), dim3(256), 0, stream>>>(hidden, projF, lnw, lnb, out);
}
